// LinearAttentionLayer_55551107006697
// MI455X (gfx1250) — matrix operand order repaired and checked by reading, not compiled, not run on hardware
//
#include <hip/hip_runtime.h>
#include <hip/hip_bf16.h>
#include <math.h>

// ---------------------------------------------------------------------------
// Types / WMMA helper
// ---------------------------------------------------------------------------
typedef __bf16 bf16_t;
typedef __attribute__((ext_vector_type(16))) __bf16 v16bf;
typedef __attribute__((ext_vector_type(8)))  __bf16 v8bf;
typedef __attribute__((ext_vector_type(8)))  float  v8f;

static __device__ __forceinline__ v8f wmma_bf16(v16bf a, v16bf b, v8f c) {
  return __builtin_amdgcn_wmma_f32_16x16x32_bf16(false, a, false, b, (short)0, c,
                                                 false, false);
}

static __device__ __forceinline__ v8f vzero8() {
  v8f z;
#pragma unroll
  for (int i = 0; i < 8; ++i) z[i] = 0.0f;
  return z;
}

// CDNA5 async global->LDS copy (16B per lane), tracked by ASYNCcnt.
// lds32 = low 32 bits of generic shared pointer (== LDS byte offset).
#define ASYNC_LDS_B128(lds32, gaddr64)                                       \
  asm volatile("global_load_async_to_lds_b128 %0, %1, off" ::"v"(lds32),     \
               "v"(gaddr64)                                                  \
               : "memory")
// Steady-state wait: previous tile's 6-op group retired, next group in flight.
#define WAIT_ASYNC_LE6() asm volatile("s_wait_asynccnt 0x6" ::: "memory")
#define WAIT_ASYNC_LE0() asm volatile("s_wait_asynccnt 0x0" ::: "memory")

static __device__ __forceinline__ unsigned lds_addr32(const void* p) {
  return (unsigned)(unsigned long long)p;
}

// A fragment (16x32 bf16), source tile row-major in K. Lane l (wave32):
//   m = m0 + (l&15); halves k0+base+0..7 and k0+base+16..23, base=(l>=16)?8:0
static __device__ __forceinline__ v16bf ldsA(const bf16_t* s, int ld, int m0, int k0) {
  const int l = threadIdx.x & 31;
  const bf16_t* p = s + (m0 + (l & 15)) * ld + k0 + ((l >> 4) << 3);
  v8bf lo = *(const v8bf*)(p);
  v8bf hi = *(const v8bf*)(p + 16);
  v16bf r;
#pragma unroll
  for (int i = 0; i < 8; ++i) { r[i] = lo[i]; r[i + 8] = hi[i]; }
  return r;
}

// B fragment (32x16 bf16), source tile stored TRANSPOSED (N-major, row = column
// of B, contiguous in K). Lane l: n = n0 + (l&15); halves k0+base+0..7 and k0+base+16..23, base=(l>=16)?8:0
static __device__ __forceinline__ v16bf ldsB(const bf16_t* s, int ld, int n0, int k0) {
  const int l = threadIdx.x & 31;
  const bf16_t* p = s + (n0 + (l & 15)) * ld + k0 + ((l >> 4) << 3);
  v8bf lo = *(const v8bf*)(p);
  v8bf hi = *(const v8bf*)(p + 16);
  v16bf r;
#pragma unroll
  for (int i = 0; i < 8; ++i) { r[i] = lo[i]; r[i + 8] = hi[i]; }
  return r;
}

static __device__ __forceinline__ float feature_map(float x) {
  // elu(x)+1 :  x>0 -> x+1 ; else exp(x)
  return x > 0.0f ? x + 1.0f : __expf(x);
}

// ---------------------------------------------------------------------------
// Conversion / transpose kernels (f32 -> bf16)
// ---------------------------------------------------------------------------
__global__ void k_cvt_bf16(const float* __restrict__ in, bf16_t* __restrict__ out,
                           int n) {
  int i = blockIdx.x * 256 + threadIdx.x;
  if (i < n) out[i] = (bf16_t)in[i];
}

// in: (K x N) f32 row-major -> out: (N x K) bf16 row-major (i.e. B transposed)
__global__ void k_transpose_cvt(const float* __restrict__ in, bf16_t* __restrict__ out,
                                int K, int N) {
  int idx = blockIdx.x * 256 + threadIdx.x;
  if (idx < K * N) {
    int k = idx / N;
    int c = idx - k * N;
    out[c * K + k] = (bf16_t)in[idx];
  }
}

// ---------------------------------------------------------------------------
// Double-buffered async-staged WMMA GEMM core (tile 128x64, BK=64, 8 waves).
// Steady state: issue async group for tile t+1, s_wait_asynccnt 6 (tile t
// complete, in-order), barrier, 8 WMMAs on tile t, barrier.
// ---------------------------------------------------------------------------
#define A_TILE_BYTES (128 * 72 * 2)
#define B_TILE_BYTES (64 * 72 * 2)

#define GEMM_STAGE(t)                                                         \
  do {                                                                        \
    unsigned sel_ = (unsigned)(t) & 1u;                                       \
    unsigned long long ga_ = aG0 + (unsigned long long)(t) * 128ull;          \
    unsigned long long gb_ = bG0 + (unsigned long long)(t) * 128ull;          \
    unsigned al_ = aL + sel_ * (unsigned)A_TILE_BYTES;                        \
    unsigned bl_ = bL + sel_ * (unsigned)B_TILE_BYTES;                        \
    _Pragma("unroll") for (int i_ = 0; i_ < 4; ++i_)                          \
        ASYNC_LDS_B128(al_ + i_ * 16u, ga_ + i_ * 16ull);                     \
    _Pragma("unroll") for (int i_ = 0; i_ < 2; ++i_)                          \
        ASYNC_LDS_B128(bl_ + i_ * 16u, gb_ + i_ * 16ull);                     \
  } while (0)

#define GEMM_COMPUTE(t)                                                       \
  do {                                                                        \
    const bf16_t* Asb_ = As[(t) & 1];                                         \
    const bf16_t* Bsb_ = Bs[(t) & 1];                                         \
    _Pragma("unroll") for (int ks = 0; ks < 2; ++ks) {                        \
      v16bf a0 = ldsA(Asb_, 72, wm * 32 + 0, ks * 32);                        \
      v16bf a1 = ldsA(Asb_, 72, wm * 32 + 16, ks * 32);                       \
      v16bf b0 = ldsB(Bsb_, 72, wn * 32 + 0, ks * 32);                        \
      v16bf b1 = ldsB(Bsb_, 72, wn * 32 + 16, ks * 32);                       \
      acc[0][0] = wmma_bf16(a0, b0, acc[0][0]);                               \
      acc[0][1] = wmma_bf16(a0, b1, acc[0][1]);                               \
      acc[1][0] = wmma_bf16(a1, b0, acc[1][0]);                               \
      acc[1][1] = wmma_bf16(a1, b1, acc[1][1]);                               \
    }                                                                         \
  } while (0)

// ---------------------------------------------------------------------------
// GEMM1: qkv = Xb(4096x1024) @ Wqkv (transposed bf16 copy), fused elu+1 on
// q/k, scatter into per-head (b,h,n,d) bf16 Q/K/V buffers.
// ---------------------------------------------------------------------------
__global__ __launch_bounds__(256) void k_gemm_qkv(
    const bf16_t* __restrict__ X, const bf16_t* __restrict__ Wt,
    bf16_t* __restrict__ Qb, bf16_t* __restrict__ Kb, bf16_t* __restrict__ Vb) {
  __shared__ bf16_t As[2][128 * 72];
  __shared__ bf16_t Bs[2][64 * 72];
  const int tid = threadIdx.x;
  const int lane = tid & 31, wave = tid >> 5;
  const int wm = wave & 3, wn = wave >> 2;
  const int M0 = blockIdx.y * 128, N0 = blockIdx.x * 64;

  const int arow = tid >> 1, aoff = (tid & 1) * 32;
  const int brow = tid >> 2, boff = (tid & 3) * 16;
  const unsigned aL = lds_addr32(As[0] + arow * 72 + aoff);
  const unsigned bL = lds_addr32(Bs[0] + brow * 72 + boff);
  const unsigned long long aG0 = (unsigned long long)(X + (M0 + arow) * 1024 + aoff);
  const unsigned long long bG0 = (unsigned long long)(Wt + (N0 + brow) * 1024 + boff);

  v8f acc[2][2];
#pragma unroll
  for (int i = 0; i < 2; ++i)
#pragma unroll
    for (int j = 0; j < 2; ++j) acc[i][j] = vzero8();

  const int T = 16;  // 1024 / 64 k-tiles
  GEMM_STAGE(0);
  for (int t = 0; t < T - 1; ++t) {
    GEMM_STAGE(t + 1);
    WAIT_ASYNC_LE6();
    __syncthreads();
    GEMM_COMPUTE(t);
    __syncthreads();
  }
  WAIT_ASYNC_LE0();
  __syncthreads();
  GEMM_COMPUTE(T - 1);

  // Epilogue: column c in [0,1536): 0..511 q, 512..1023 k, 1024..1535 v.
  const int jlane = lane & 15;
  const int rbase = (lane >> 4) * 8;
#pragma unroll
  for (int mi = 0; mi < 2; ++mi)
#pragma unroll
    for (int ni = 0; ni < 2; ++ni) {
      int c = N0 + wn * 32 + ni * 16 + jlane;
      int sec = c >> 9;
      int cc = c & 511;
      int hh = cc >> 6, dd = cc & 63;
      bf16_t* dst = (sec == 0) ? Qb : ((sec == 1) ? Kb : Vb);
#pragma unroll
      for (int r = 0; r < 8; ++r) {
        int R = M0 + wm * 32 + mi * 16 + rbase + r;
        int bb = R >> 11, nn = R & 2047;
        float v = acc[mi][ni][r];
        if (sec < 2) v = feature_map(v);
        dst[((bb * 8 + hh) * 2048 + nn) * 64 + dd] = (bf16_t)v;
      }
    }
}

// ---------------------------------------------------------------------------
// Causal linear attention, chunked (C=32) state-space form.
// One workgroup per (b,h); 4 waves, wave w owns e-columns [16w, 16w+16).
// Per chunk: Ninter = Q@S ; P = QK^T masked ; Nintra = P@V ;
//            denom = P@ones + Q@Ksum ; out = N/denom ; S += K^T V ; Ksum += K^T ones
// All matmuls via v_wmma_f32_16x16x32_bf16; P/S re-layout f32->bf16 via LDS.
// Q chunk staged via async global->LDS; K/V need transposes so stage via VGPRs.
// ---------------------------------------------------------------------------
__global__ __launch_bounds__(128) void k_attn(
    const bf16_t* __restrict__ Qb, const bf16_t* __restrict__ Kb,
    const bf16_t* __restrict__ Vb, bf16_t* __restrict__ Ob) {
  __shared__ bf16_t Qc[32 * 72];     // Q chunk, row-major (c x d)
  __shared__ bf16_t Kc[32 * 72];     // K chunk, row-major (c x d)  -> B of P
  __shared__ bf16_t Kt[64 * 40];     // K chunk transposed (d x c)  -> A of S-update
  __shared__ bf16_t Vt[64 * 40];     // V chunk transposed (e x c)  -> B frags
  __shared__ bf16_t Pa[4][32 * 40];  // per-wave masked P, bf16, row-major (i x j)
  __shared__ bf16_t St[4][16 * 72];  // per-wave S strip, (e x d), B-friendly
  __shared__ bf16_t Sdt[4][16 * 72]; // per-wave Ksum strip, (e x d)

  const int tid = threadIdx.x, lane = tid & 31, wave = tid >> 5;
  const int bh = blockIdx.x, b = bh >> 3, h = bh & 7;
  const bf16_t* Qg = Qb + bh * 2048 * 64;
  const bf16_t* Kg = Kb + bh * 2048 * 64;
  const bf16_t* Vg = Vb + bh * 2048 * 64;
  const int jlane = lane & 15;
  const int rbase = (lane >> 4) * 8;

  const int srow = tid >> 2, soff = (tid & 3) * 16;
  unsigned qL = lds_addr32(Qc + srow * 72 + soff);
  unsigned long long qG = (unsigned long long)(Qg + srow * 64 + soff);

  // zero this wave's state strips
  for (int i = lane; i < 16 * 72; i += 32) {
    St[wave][i] = (bf16_t)0.0f;
    Sdt[wave][i] = (bf16_t)0.0f;
  }
  v8f Sacc[4], Sdacc[4];
#pragma unroll
  for (int t = 0; t < 4; ++t) { Sacc[t] = vzero8(); Sdacc[t] = vzero8(); }

  v16bf ones;
#pragma unroll
  for (int i = 0; i < 16; ++i) ones[i] = (bf16_t)1.0f;

  for (int c0 = 0; c0 < 2048; c0 += 32) {
    __syncthreads();
    // Q chunk: pure copy -> async global->LDS (32B per thread)
    ASYNC_LDS_B128(qL, qG);
    ASYNC_LDS_B128(qL + 16u, qG + 16ull);
    qG += 32ull * 64ull * 2ull;  // next chunk (32 rows x 64 halfs)
    {  // K/V chunks need transposed copies -> VGPR round trip
      const bf16_t* gk = Kg + (c0 + srow) * 64 + soff;
      v8bf k0v = *(const v8bf*)(gk);
      v8bf k1v = *(const v8bf*)(gk + 8);
      *(v8bf*)(Kc + srow * 72 + soff) = k0v;
      *(v8bf*)(Kc + srow * 72 + soff + 8) = k1v;
#pragma unroll
      for (int i = 0; i < 8; ++i) {
        Kt[(soff + i) * 40 + srow] = k0v[i];
        Kt[(soff + 8 + i) * 40 + srow] = k1v[i];
      }
      const bf16_t* gv = Vg + (c0 + srow) * 64 + soff;
      v8bf v0v = *(const v8bf*)(gv);
      v8bf v1v = *(const v8bf*)(gv + 8);
#pragma unroll
      for (int i = 0; i < 8; ++i) {
        Vt[(soff + i) * 40 + srow] = v0v[i];
        Vt[(soff + 8 + i) * 40 + srow] = v1v[i];
      }
    }
    WAIT_ASYNC_LE0();
    __syncthreads();

    // Q fragments: 2 row-tiles x 2 k-steps over d=64
    v16bf qa[2][2];
#pragma unroll
    for (int mi = 0; mi < 2; ++mi)
#pragma unroll
      for (int ks = 0; ks < 2; ++ks) qa[mi][ks] = ldsA(Qc, 72, mi * 16, ks * 32);

    // Inter-chunk numerator (Q@S) and denominator (Q@Ksum)
    v8f nin[2] = {vzero8(), vzero8()};
    v8f dac[2] = {vzero8(), vzero8()};
#pragma unroll
    for (int ks = 0; ks < 2; ++ks) {
      v16bf bS = ldsB(St[wave], 72, 0, ks * 32);
      v16bf bSd = ldsB(Sdt[wave], 72, 0, ks * 32);
      nin[0] = wmma_bf16(qa[0][ks], bS, nin[0]);
      nin[1] = wmma_bf16(qa[1][ks], bS, nin[1]);
      dac[0] = wmma_bf16(qa[0][ks], bSd, dac[0]);
      dac[1] = wmma_bf16(qa[1][ks], bSd, dac[1]);
    }

    // P = Q @ K^T  (32x32), contraction over d=64
    v8f p[2][2];
#pragma unroll
    for (int i = 0; i < 2; ++i)
#pragma unroll
      for (int j = 0; j < 2; ++j) p[i][j] = vzero8();
#pragma unroll
    for (int ks = 0; ks < 2; ++ks) {
      v16bf bk0 = ldsB(Kc, 72, 0, ks * 32);
      v16bf bk1 = ldsB(Kc, 72, 16, ks * 32);
      p[0][0] = wmma_bf16(qa[0][ks], bk0, p[0][0]);
      p[0][1] = wmma_bf16(qa[0][ks], bk1, p[0][1]);
      p[1][0] = wmma_bf16(qa[1][ks], bk0, p[1][0]);
      p[1][1] = wmma_bf16(qa[1][ks], bk1, p[1][1]);
    }

    // Causal mask + f32->bf16 relayout of P into A-friendly LDS tile
#pragma unroll
    for (int mi = 0; mi < 2; ++mi)
#pragma unroll
      for (int ni = 0; ni < 2; ++ni)
#pragma unroll
        for (int r = 0; r < 8; ++r) {
          int ii = mi * 16 + rbase + r;
          int jj = ni * 16 + jlane;
          float v = (jj <= ii) ? p[mi][ni][r] : 0.0f;
          Pa[wave][ii * 40 + jj] = (bf16_t)v;
        }

    v16bf paf0 = ldsA(Pa[wave], 40, 0, 0);
    v16bf paf1 = ldsA(Pa[wave], 40, 16, 0);
    v16bf bV = ldsB(Vt, 40, wave * 16, 0);

    // Intra-chunk numerator and denominator row-sums
    nin[0] = wmma_bf16(paf0, bV, nin[0]);
    nin[1] = wmma_bf16(paf1, bV, nin[1]);
    dac[0] = wmma_bf16(paf0, ones, dac[0]);
    dac[1] = wmma_bf16(paf1, ones, dac[1]);

    // Output strip: (b, n, h*64 + e) in bf16 O buffer (4096 x 512)
#pragma unroll
    for (int mi = 0; mi < 2; ++mi)
#pragma unroll
      for (int r = 0; r < 8; ++r) {
        int ii = mi * 16 + rbase + r;
        int row = b * 2048 + c0 + ii;
        int col = h * 64 + wave * 16 + jlane;
        float o = nin[mi][r] / (dac[mi][r] + 1e-6f);
        Ob[row * 512 + col] = (bf16_t)o;
      }

    // State update: S += K^T V ; Ksum += K^T ones   (4 d-row tiles)
#pragma unroll
    for (int t = 0; t < 4; ++t) {
      v16bf ka = ldsA(Kt, 40, t * 16, 0);
      Sacc[t] = wmma_bf16(ka, bV, Sacc[t]);
      Sdacc[t] = wmma_bf16(ka, ones, Sdacc[t]);
    }

    // Re-layout state to bf16 (e x d) strips for next chunk's B fragments
#pragma unroll
    for (int t = 0; t < 4; ++t)
#pragma unroll
      for (int r = 0; r < 8; ++r) {
        int dd = t * 16 + rbase + r;
        St[wave][jlane * 72 + dd] = (bf16_t)Sacc[t][r];
        Sdt[wave][jlane * 72 + dd] = (bf16_t)Sdacc[t][r];
      }
  }
}

// ---------------------------------------------------------------------------
// GEMM2: out = O(4096x512) @ W_out + b_out, f32 output.
// ---------------------------------------------------------------------------
__global__ __launch_bounds__(256) void k_gemm_out(
    const bf16_t* __restrict__ O, const bf16_t* __restrict__ Wt,
    const float* __restrict__ bias, float* __restrict__ out) {
  __shared__ bf16_t As[2][128 * 72];
  __shared__ bf16_t Bs[2][64 * 72];
  const int tid = threadIdx.x;
  const int lane = tid & 31, wave = tid >> 5;
  const int wm = wave & 3, wn = wave >> 2;
  const int M0 = blockIdx.y * 128, N0 = blockIdx.x * 64;

  const int arow = tid >> 1, aoff = (tid & 1) * 32;
  const int brow = tid >> 2, boff = (tid & 3) * 16;
  const unsigned aL = lds_addr32(As[0] + arow * 72 + aoff);
  const unsigned bL = lds_addr32(Bs[0] + brow * 72 + boff);
  const unsigned long long aG0 = (unsigned long long)(O + (M0 + arow) * 512 + aoff);
  const unsigned long long bG0 = (unsigned long long)(Wt + (N0 + brow) * 512 + boff);

  v8f acc[2][2];
#pragma unroll
  for (int i = 0; i < 2; ++i)
#pragma unroll
    for (int j = 0; j < 2; ++j) acc[i][j] = vzero8();

  const int T = 8;  // 512 / 64 k-tiles
  GEMM_STAGE(0);
  for (int t = 0; t < T - 1; ++t) {
    GEMM_STAGE(t + 1);
    WAIT_ASYNC_LE6();
    __syncthreads();
    GEMM_COMPUTE(t);
    __syncthreads();
  }
  WAIT_ASYNC_LE0();
  __syncthreads();
  GEMM_COMPUTE(T - 1);

  const int jlane = lane & 15;
  const int rbase = (lane >> 4) * 8;
#pragma unroll
  for (int mi = 0; mi < 2; ++mi)
#pragma unroll
    for (int ni = 0; ni < 2; ++ni) {
      int c = N0 + wn * 32 + ni * 16 + jlane;
      float bia = bias[c];
#pragma unroll
      for (int r = 0; r < 8; ++r) {
        int R = M0 + wm * 32 + mi * 16 + rbase + r;
        out[R * 1024 + c] = acc[mi][ni][r] + bia;
      }
    }
}

// ---------------------------------------------------------------------------
// Launch
// ---------------------------------------------------------------------------
extern "C" void kernel_launch(void* const* d_in, const int* in_sizes, int n_in,
                              void* d_out, int out_size, void* d_ws, size_t ws_size,
                              hipStream_t stream) {
  const float* x = (const float*)d_in[0];      // (2,2048,1024)
  const float* Wqkv = (const float*)d_in[1];   // (1024,1536)
  const float* Wout = (const float*)d_in[2];   // (512,1024)
  const float* bout = (const float*)d_in[3];   // (1024,)
  float* out = (float*)d_out;                  // (2,2048,1024)

  char* ws = (char*)d_ws;
  bf16_t* Xb    = (bf16_t*)(ws + 0);           // 4096*1024 bf16 =  8 MB
  bf16_t* Wqkvt = (bf16_t*)(ws + 8388608);     // 1536*1024 bf16 =  3 MB
  bf16_t* Woutt = (bf16_t*)(ws + 11534336);    // 1024*512  bf16 =  1 MB
  bf16_t* Qb    = (bf16_t*)(ws + 12582912);    // 4096*512  bf16 =  4 MB
  bf16_t* Kb    = (bf16_t*)(ws + 16777216);    // 4 MB
  bf16_t* Vb    = (bf16_t*)(ws + 20971520);    // 4 MB
  bf16_t* Ob    = (bf16_t*)(ws + 25165824);    // 4 MB

  k_cvt_bf16<<<(4096 * 1024 + 255) / 256, 256, 0, stream>>>(x, Xb, 4096 * 1024);
  k_transpose_cvt<<<(1024 * 1536 + 255) / 256, 256, 0, stream>>>(Wqkv, Wqkvt, 1024, 1536);
  k_transpose_cvt<<<(512 * 1024 + 255) / 256, 256, 0, stream>>>(Wout, Woutt, 512, 1024);

  k_gemm_qkv<<<dim3(24, 32), 256, 0, stream>>>(Xb, Wqkvt, Qb, Kb, Vb);
  k_attn<<<16, 128, 0, stream>>>(Qb, Kb, Vb, Ob);
  k_gemm_out<<<dim3(16, 32), 256, 0, stream>>>(Ob, Woutt, bout, out);
}
